// SpecificMoE_23012434772537
// MI455X (gfx1250) — compile-verified, matrix operand order checked by reading, not run on hardware
//
#include <hip/hip_runtime.h>
#include <math.h>

// ---------------- Problem constants ----------------
#define BB 4
#define SS 2048
#define HH 512
#define FF 2048
#define EE 8
#define NT (BB*SS)        // 8192 tokens
#define NP (NT*2)         // 16384 (token,expert) pairs
#define PCAP 17408        // pairs padded to 128 per expert, worst case
#define WGMAX 136         // max GEMM workgroups (128 rows each)

// ---------------- Workspace layout (bytes) ----------------
#define OFF_COUNTS 0u           // 8 u32
#define OFF_BASE   256u         // 8 i32
#define OFF_WGE    512u         // 136 i32
#define OFF_WGPB   1536u        // 136 i32
#define OFF_TOKE   4096u        // NP i32
#define OFF_TOKS   (OFF_TOKE + 65536u)
#define OFF_GATE   (OFF_TOKS + 65536u)
#define OFF_PROW   (OFF_GATE + 65536u)
#define OFF_FC1P   (OFF_PROW + 65536u)          // 16 MiB bf16 frags
#define OFF_FC2P   (OFF_FC1P + 16777216u)       // 16 MiB bf16 frags
#define OFF_XG     (OFF_FC2P + 16777216u)       // PCAP*512*2 = 17825792
#define OFF_Y      (OFF_XG + 17825792u)         // PCAP*512*4 = 35651584

typedef __attribute__((ext_vector_type(16))) __bf16 v16bf;
typedef __attribute__((ext_vector_type(8)))  float  v8f;

struct Frag32B { uint4 a, b; };   // 32 bytes = one lane's v16bf

__device__ __forceinline__ v16bf ld_frag(const uint4* p) {
  Frag32B t; t.a = p[0]; t.b = p[1];
  return __builtin_bit_cast(v16bf, t);
}

__device__ __forceinline__ v8f vzero8() {
  v8f z;
  #pragma unroll
  for (int i = 0; i < 8; ++i) z[i] = 0.0f;
  return z;
}

__device__ __forceinline__ unsigned f2bf(float f) {
  union { float f; unsigned u; } v; v.f = f;
  unsigned u = v.u;
  return ((u + 0x7fffu + ((u >> 16) & 1u)) >> 16) & 0xffffu;  // RNE
}

// pack 8 contiguous floats -> 8 bf16 in one uint4 (16 bytes)
__device__ __forceinline__ uint4 pack8(const float* p) {
  uint4 r;
  r.x = f2bf(p[0]) | (f2bf(p[1]) << 16);
  r.y = f2bf(p[2]) | (f2bf(p[3]) << 16);
  r.z = f2bf(p[4]) | (f2bf(p[5]) << 16);
  r.w = f2bf(p[6]) | (f2bf(p[7]) << 16);
  return r;
}

__device__ __forceinline__ float gelu_exact(float x) {
  return 0.5f * x * (1.0f + erff(x * 0.70710678118654752f));
}

__device__ __forceinline__ v8f wmma_bf16(v16bf a, v16bf b, v8f c) {
  return __builtin_amdgcn_wmma_f32_16x16x32_bf16(false, a, false, b,
                                                 (short)0, c, false, false);
}

// ---------------- 1) zero counters + gathered-x buffer ----------------
__global__ void k_zero(uint4* hdr, uint4* xg) {
  int gid = blockIdx.x * 256 + threadIdx.x;
  uint4 z; z.x = z.y = z.z = z.w = 0u;
  if (gid < 256) hdr[gid] = z;
  int i = gid - 256;
  if (i >= 0 && i < (int)(17825792u / 16u)) xg[i] = z;
}

// ---------------- 2) pack fc1/fc2 weights to bf16 WMMA fragments ------
// A-layout of a 16x32 tile: frag[lane][j] = M[lane%16][koff(lane,j)]
// koff = lane<16 ? (j<8 ? j : j+8) : (j<8 ? j+8 : j+16)
__global__ void k_pack(const float* __restrict__ fc1_w,
                       const float* __restrict__ fc2_w,
                       uint4* __restrict__ fc1p, uint4* __restrict__ fc2p) {
  int gid = blockIdx.x * 256 + threadIdx.x;          // 1048576 threads
  int lane = gid & 31;
  int frag = (gid >> 5) & 16383;
  int off  = (lane >> 4) << 3;                       // 0 or 8
  if (gid < (1 << 19)) {
    // fc1: frag = ((e*64+fc)*32 + nt*16 + kc); tile rows = f, K = h
    int kc = frag & 15, nt = (frag >> 4) & 1, fc = (frag >> 5) & 63, e = frag >> 11;
    const float* src = fc1_w + ((size_t)e * FF + fc * 32 + nt * 16 + (lane & 15)) * HH + kc * 32;
    uint4* d = fc1p + (size_t)frag * 64 + lane * 2;
    d[0] = pack8(src + off);
    d[1] = pack8(src + 16 + off);
  } else {
    // fc2: frag = ((e*64+fc)*32 + nt2); tile rows = h, K = f
    int nt = frag & 31, fc = (frag >> 5) & 63, e = frag >> 11;
    const float* src = fc2_w + ((size_t)e * HH + nt * 16 + (lane & 15)) * FF + fc * 32;
    uint4* d = fc2p + (size_t)frag * 64 + lane * 2;
    d[0] = pack8(src + off);
    d[1] = pack8(src + 16 + off);
  }
}

// ---------------- 3) router: logits, softmax, top-2 -------------------
__global__ void k_router(const float* __restrict__ x, const float* __restrict__ rw,
                         float* __restrict__ probs_out, float* __restrict__ topk_out,
                         unsigned* __restrict__ counts, int* __restrict__ tok_e,
                         int* __restrict__ tok_s, float* __restrict__ gate) {
  int t = blockIdx.x * 8 + (threadIdx.x >> 5);
  int lane = threadIdx.x & 31;
  const float* xp = x + (size_t)t * HH;
  float acc[EE];
  #pragma unroll
  for (int e = 0; e < EE; ++e) acc[e] = 0.0f;
  #pragma unroll 4
  for (int i = 0; i < 16; ++i) {
    int h = lane + 32 * i;
    float xv = xp[h];
    #pragma unroll
    for (int e = 0; e < EE; ++e) acc[e] += xv * rw[e * HH + h];
  }
  #pragma unroll
  for (int m = 16; m >= 1; m >>= 1)
    #pragma unroll
    for (int e = 0; e < EE; ++e) acc[e] += __shfl_xor(acc[e], m, 32);

  if (lane == 0) {
    float mx = acc[0];
    #pragma unroll
    for (int e = 1; e < EE; ++e) mx = fmaxf(mx, acc[e]);
    float p[EE], s = 0.0f;
    #pragma unroll
    for (int e = 0; e < EE; ++e) { p[e] = expf(acc[e] - mx); s += p[e]; }
    float inv = 1.0f / s;
    #pragma unroll
    for (int e = 0; e < EE; ++e) { p[e] *= inv; probs_out[t * EE + e] = p[e]; }
    int i0 = 0;
    #pragma unroll
    for (int e = 1; e < EE; ++e) if (p[e] > p[i0]) i0 = e;   // first max (tie -> lower idx)
    int i1 = (i0 == 0) ? 1 : 0;
    #pragma unroll
    for (int e = 0; e < EE; ++e) if (e != i1 && e != i0 && p[e] > p[i1]) i1 = e;
    float s2 = p[i0] + p[i1] + 1e-9f;
    float w0 = p[i0] / s2, w1 = p[i1] / s2;
    topk_out[t * 2 + 0] = (float)i0;
    topk_out[t * 2 + 1] = (float)i1;
    unsigned s0 = atomicAdd(&counts[i0], 1u);
    unsigned s1 = atomicAdd(&counts[i1], 1u);
    tok_e[t * 2 + 0] = i0; tok_s[t * 2 + 0] = (int)s0; gate[t * 2 + 0] = w0;
    tok_e[t * 2 + 1] = i1; tok_s[t * 2 + 1] = (int)s1; gate[t * 2 + 1] = w1;
  }
}

// ---------------- 4) prefix: expert bases + WG table -------------------
__global__ void k_prefix(const unsigned* __restrict__ counts, int* __restrict__ base,
                         int* __restrict__ wg_e, int* __restrict__ wg_pb) {
  if (threadIdx.x == 0 && blockIdx.x == 0) {
    int b = 0, idx = 0;
    for (int e = 0; e < EE; ++e) {
      int c = (int)counts[e];
      int pc = (c + 127) & ~127;           // pad rows to 128 (WG granularity)
      base[e] = b;
      int nw = pc >> 7;
      for (int w = 0; w < nw; ++w) { wg_e[idx] = e; wg_pb[idx] = b + (w << 7); ++idx; }
      b += pc;
    }
    for (; idx < WGMAX; ++idx) { wg_e[idx] = -1; wg_pb[idx] = 0; }
  }
}

// ---------------- 5) scatter x rows into A-fragment order --------------
__global__ void k_scatter(const float* __restrict__ x, const int* __restrict__ tok_e,
                          const int* __restrict__ tok_s, const int* __restrict__ base,
                          int* __restrict__ prow, uint4* __restrict__ xg) {
  int gid = blockIdx.x * 256 + threadIdx.x;   // NP*16 threads
  int pid = gid >> 4, kc = gid & 15;
  int t = pid >> 1;
  int e = tok_e[pid], slot = tok_s[pid];
  int row = base[e] + slot;
  if (kc == 0) prow[pid] = row;
  int m = row & 15, tile = row >> 4;
  const float* src = x + (size_t)t * HH + kc * 32;
  uint4* dst = xg + (size_t)tile * 1024 + kc * 64;   // tile*16KB in uint4
  // lane m   (off=0): K {0..7},{16..23} ; lane m+16 (off=8): K {8..15},{24..31}
  dst[m * 2 + 0]        = pack8(src + 0);
  dst[m * 2 + 1]        = pack8(src + 16);
  dst[(m + 16) * 2 + 0] = pack8(src + 8);
  dst[(m + 16) * 2 + 1] = pack8(src + 24);
}

// ---------------- 6) fused FFN: y = gelu(x@fc1^T + b1) @ fc2^T ---------
// WG = 128 rows of one expert; 8 waves, wave w owns 16-row tile, 512 cols.
// Stage 1 computed transposed (D = fc1_tile x x^T) so its C registers are
// directly the stage-2 A fragment after in-lane bias+GELU+bf16 convert.
// Weight staging is double-buffered in LDS (2 x 64 KB) with register
// prefetch: global loads for chunk fc+1 are issued before computing chunk
// fc, hiding all weight traffic behind the 64 WMMAs; one barrier/chunk.
__global__ void __launch_bounds__(256, 1)
k_gemm(const uint4* __restrict__ fc1p, const uint4* __restrict__ fc2p,
       const uint4* __restrict__ xg, const float* __restrict__ fc1_b,
       float* __restrict__ y, const int* __restrict__ wg_e,
       const int* __restrict__ wg_pb) {
  __shared__ uint4 sX[8192];       // 128 KB: 8 tiles x 16 k-chunks x 1KB frags
  __shared__ uint4 sW[2][4096];    // 2 x 64 KB: fc1 [0..2047], fc2 [2048..4095]

  int e = wg_e[blockIdx.x];
  if (e < 0) return;
  int pb = wg_pb[blockIdx.x];
  int tid = threadIdx.x, lane = tid & 31, wave = tid >> 5;
  int off = (lane >> 4) << 3;                 // 0 or 8

  // stage gathered x fragments (once; covered by first loop barrier)
  const uint4* xsrc = xg + (size_t)pb * 64;   // pb*1024 bytes
  for (int i = tid; i < 8192; i += 256) sX[i] = xsrc[i];

  // register-prefetch chunk 0 (16 uint4 per thread = 64 KB per WG)
  const uint4* wbase1 = fc1p + (size_t)e * 64 * 2048;
  const uint4* wbase2 = fc2p + (size_t)e * 64 * 2048;
  uint4 r1[8], r2[8];
  #pragma unroll
  for (int i = 0; i < 8; ++i) { r1[i] = wbase1[tid + i * 256]; r2[i] = wbase2[tid + i * 256]; }

  v8f acc[32];
  #pragma unroll
  for (int n = 0; n < 32; ++n) acc[n] = vzero8();

  const float* b1 = fc1_b + e * FF;
  const uint4* xw = &sX[wave * 1024 + lane * 2];

  #pragma unroll 1
  for (int fc = 0; fc < 64; ++fc) {
    // dump prefetched chunk into the inactive buffer
    uint4* wb = &sW[fc & 1][0];
    #pragma unroll
    for (int i = 0; i < 8; ++i) {
      wb[tid + i * 256]        = r1[i];
      wb[2048 + tid + i * 256] = r2[i];
    }
    // issue global loads for the next chunk; they complete during compute
    if (fc < 63) {
      const uint4* n1 = wbase1 + (size_t)(fc + 1) * 2048;
      const uint4* n2 = wbase2 + (size_t)(fc + 1) * 2048;
      #pragma unroll
      for (int i = 0; i < 8; ++i) { r1[i] = n1[tid + i * 256]; r2[i] = n2[tid + i * 256]; }
    }
    __syncthreads();   // staging of current buffer visible to all waves

    // stage 1: h1^T tiles (f x tok), K = H in 16 chunks of 32
    v8f c1a = vzero8(), c1b = vzero8();
    #pragma unroll
    for (int kc = 0; kc < 16; ++kc) {
      v16bf xf = ld_frag(xw + kc * 64);
      v16bf a0 = ld_frag(&wb[kc * 64 + lane * 2]);
      c1a = wmma_bf16(a0, xf, c1a);
      v16bf a1 = ld_frag(&wb[(16 + kc) * 64 + lane * 2]);
      c1b = wmma_bf16(a1, xf, c1b);
    }

    // bias + exact GELU + bf16 -> stage-2 A fragment (pure in-lane)
    const float* bb = b1 + fc * 32;
    v16bf ap;
    #pragma unroll
    for (int i = 0; i < 8; ++i) {
      float v0 = c1a[i] + bb[off + i];
      float v1 = c1b[i] + bb[16 + off + i];
      ap[i]     = (__bf16)gelu_exact(v0);
      ap[i + 8] = (__bf16)gelu_exact(v1);
    }

    // stage 2: y(tok x h) += a(tok x 32f) @ fc2^T, 32 column tiles
    #pragma unroll
    for (int nt = 0; nt < 32; ++nt) {
      v16bf wB = ld_frag(&wb[2048 + nt * 64 + lane * 2]);
      acc[nt] = wmma_bf16(ap, wB, acc[nt]);
    }
  }

  // store y (no bias; combine adds fc2_b)
  float* yrow = y + (size_t)(pb + wave * 16 + off) * HH + (lane & 15);
  #pragma unroll
  for (int nt = 0; nt < 32; ++nt)
    #pragma unroll
    for (int i = 0; i < 8; ++i)
      yrow[(size_t)i * HH + nt * 16] = acc[nt][i];
}

// ---------------- 7) combine: out = sum_k gate*(y_row + fc2_b) ---------
__global__ void k_combine(const float* __restrict__ y, const float* __restrict__ fc2_b,
                          const int* __restrict__ prow, const int* __restrict__ tok_e,
                          const float* __restrict__ gate, float* __restrict__ out) {
  int t = blockIdx.x, tid = threadIdx.x;
  int e0 = tok_e[2 * t], e1 = tok_e[2 * t + 1];
  int r0 = prow[2 * t], r1 = prow[2 * t + 1];
  float w0 = gate[2 * t], w1 = gate[2 * t + 1];
  #pragma unroll
  for (int j = 0; j < 2; ++j) {
    int h = tid + j * 256;
    float v0 = y[(size_t)r0 * HH + h] + fc2_b[e0 * HH + h];
    float v1 = y[(size_t)r1 * HH + h] + fc2_b[e1 * HH + h];
    out[(size_t)t * HH + h] = w0 * v0 + w1 * v1;
  }
}

// ---------------- host launcher ----------------------------------------
extern "C" void kernel_launch(void* const* d_in, const int* in_sizes, int n_in,
                              void* d_out, int out_size, void* d_ws, size_t ws_size,
                              hipStream_t stream) {
  const float* x      = (const float*)d_in[0];
  const float* rw     = (const float*)d_in[1];
  const float* fc1_w  = (const float*)d_in[2];
  const float* fc1_b  = (const float*)d_in[3];
  const float* fc2_w  = (const float*)d_in[4];
  const float* fc2_b  = (const float*)d_in[5];

  char* ws = (char*)d_ws;
  unsigned* counts = (unsigned*)(ws + OFF_COUNTS);
  int*   base   = (int*)(ws + OFF_BASE);
  int*   wg_e   = (int*)(ws + OFF_WGE);
  int*   wg_pb  = (int*)(ws + OFF_WGPB);
  int*   tok_e  = (int*)(ws + OFF_TOKE);
  int*   tok_s  = (int*)(ws + OFF_TOKS);
  float* gate   = (float*)(ws + OFF_GATE);
  int*   prow   = (int*)(ws + OFF_PROW);
  uint4* fc1p   = (uint4*)(ws + OFF_FC1P);
  uint4* fc2p   = (uint4*)(ws + OFF_FC2P);
  uint4* xg     = (uint4*)(ws + OFF_XG);
  float* ybuf   = (float*)(ws + OFF_Y);

  float* out   = (float*)d_out;
  float* probs = out + (size_t)NT * HH;          // [B,S,E]
  float* topk  = probs + (size_t)NT * EE;        // [B,S,K] as floats

  k_zero<<<4353, 256, 0, stream>>>((uint4*)ws, xg);
  k_pack<<<4096, 256, 0, stream>>>(fc1_w, fc2_w, fc1p, fc2p);
  k_router<<<NT / 8, 256, 0, stream>>>(x, rw, probs, topk, counts, tok_e, tok_s, gate);
  k_prefix<<<1, 32, 0, stream>>>(counts, base, wg_e, wg_pb);
  k_scatter<<<NP * 16 / 256, 256, 0, stream>>>(x, tok_e, tok_s, base, prow, xg);
  k_gemm<<<WGMAX, 256, 0, stream>>>(fc1p, fc2p, xg, fc1_b, ybuf, wg_e, wg_pb);
  k_combine<<<NT, 256, 0, stream>>>(ybuf, fc2_b, prow, tok_e, gate, out);
}
